// DeepConvVQVAE_52209622450486
// MI455X (gfx1250) — matrix operand order repaired and checked by reading, not compiled, not run on hardware
//
#include <hip/hip_runtime.h>
#include <hip/hip_bf16.h>
#include <math.h>

// ---------------------------------------------------------------------------
// CDNA5 (gfx1250) VQ-VAE forward. All convs (incl. transposed) run through a
// single implicit-GEMM kernel template using v_wmma_f32_16x16x32_bf16 (wave32).
// Weights are pre-converted once per conv into a zero-padded bf16 [OCr][Kr]
// GEMM layout (transpose-conv flip folded in), so the A-tile staging is one
// global_load_b128 + one ds_store_b128 per thread per K-step. Block tile
// 64(M) x 128(N) per image (blockIdx.z = batch), K-step 32; 8 waves, each
// owning a 32x32 sub-tile -> 4 WMMAs per wave per K-step. Fragments are read
// from LDS as ds_load_b128. All index math is shift/mask or compile-time-
// constant division.
// ---------------------------------------------------------------------------

typedef __attribute__((ext_vector_type(16))) __bf16 v16bf;
typedef __attribute__((ext_vector_type(8)))  float  v8f;

union FragBF {
  v16bf v;
  unsigned short u[16];
  uint4 q[2];
};

__device__ __forceinline__ unsigned short f32_to_bf16_bits(float f) {
  unsigned int x = __float_as_uint(f);
  unsigned int r = x + 0x7fffu + ((x >> 16) & 1u);   // round-to-nearest-even
  return (unsigned short)(r >> 16);
}

struct ConvP {
  const float* in;            // [B, IC, H, W] f32
  const unsigned short* wb;   // [OCr][Kr] bf16, zero-padded, flip pre-applied
  float* out;                 // [B, OC, OH, OW]
  int IC, H, W, OC, pad, OH, OW, owShift, Kr;   // OW == 1 << owShift
};

#define LSTR 40   // LDS row stride in halves (80 B, 16-B aligned, bank-staggered)

template <int KH, int KW, int STRIDE, int LDIL>
__global__ __launch_bounds__(256)
void conv_wmma(ConvP p) {
  constexpr int KK = KH * KW;
  __shared__ unsigned short Alds[64 * LSTR];   // [m][k]  64 x 32
  __shared__ unsigned short Blds[128 * LSTR];  // [n][k] 128 x 32 (transposed store)

  const int K    = p.IC * KK;                  // true K (B-side bound)
  const int Kr   = p.Kr;                       // padded K (loop bound, A layout)
  const int HWo  = p.OH * p.OW;
  const int pBlk = blockIdx.x * 128;           // column within image
  const int mBlk = blockIdx.y * 64;
  const int img  = blockIdx.z;                 // batch index

  const int tid  = threadIdx.x;
  const int lane = tid & 31;
  const int wave = tid >> 5;
  const int wm   = wave >> 2;        // 0..1 -> 32-row M sub-block
  const int wn   = wave & 3;         // 0..3 -> 32-col N sub-block
  const int lr   = lane & 15;
  const int hi   = lane >> 4;        // half-wave select

  // ---- A tile staging: one b128 per thread (64 rows x 4 segments of 8 halves)
  const int arow = tid >> 2;         // 0..63
  const int aseg = (tid & 3) * 8;    // half offset within row: 0,8,16,24
  const unsigned short* wrow = p.wb + (size_t)(mBlk + arow) * Kr + aseg;

  // ---- B tile staging: column fixed per thread, k varies ----
  const int bcol = tid & 127;        // column within block (fixed)
  const int bk0  = tid >> 7;         // k-lane base; k = bk0 + 2*j, j = 0..15

  const int  gp   = pBlk + bcol;
  const bool colv = (gp < HWo);
  const int  boh  = gp >> p.owShift;           // OW is a power of two
  const int  bow  = gp & (p.OW - 1);
  const float* inb = p.in + (long)img * p.IC * p.H * p.W;

  v8f acc[2][2] = {};

  for (int k0 = 0; k0 < Kr; k0 += 32) {
    // --- A: bf16 weight tile, single vector copy per thread ---
    {
      const uint4 av = *(const uint4*)(wrow + k0);
      *(uint4*)&Alds[arow * LSTR + aseg] = av;
      if (k0 + 32 < Kr)                         // prefetch next weight segment
        __builtin_prefetch((const void*)(wrow + k0 + 32), 0, 1);
    }
    // --- B: 32x128 implicit-im2col tile (f32 -> bf16) ---
#pragma unroll
    for (int j = 0; j < 16; ++j) {
      const int kl = bk0 + 2 * j;
      const int gk = k0 + kl;
      float v = 0.f;
      if (colv && gk < K) {
        const int ic = gk / KK;      // compile-time divisor
        const int r  = gk - ic * KK;
        const int kh = r / KW;
        const int kw = r - kh * KW;
        int dh = boh * STRIDE - p.pad + kh;
        int dw = bow * STRIDE - p.pad + kw;
        bool ok = true;
        if (LDIL == 2) {                       // lhs-dilated (transposed conv)
          ok = !((dh & 1) | (dw & 1));
          dh >>= 1; dw >>= 1;
        }
        if (ok && dh >= 0 && dh < p.H && dw >= 0 && dw < p.W)
          v = inb[((long)ic * p.H + dh) * p.W + dw];
      }
      Blds[bcol * LSTR + kl] = f32_to_bf16_bits(v);
    }
    __syncthreads();

    // --- fragments (ISA 16-bit A 16x32 / B 32x16 layouts), b128 LDS reads ---
    FragBF afr[2], bfr[2];
#pragma unroll
    for (int mi = 0; mi < 2; ++mi) {
      const int row = (wm * 32 + mi * 16 + lr) * LSTR;
      afr[mi].q[0] = *(const uint4*)&Alds[row + hi * 8];        // K akb..akb+7
      afr[mi].q[1] = *(const uint4*)&Alds[row + hi * 8 + 16];   // K akb+16..akb+23
    }
#pragma unroll
    for (int ni = 0; ni < 2; ++ni) {
      const int row = (wn * 32 + ni * 16 + lr) * LSTR;
      bfr[ni].q[0] = *(const uint4*)&Blds[row + hi * 8];        // K bkb..bkb+7
      bfr[ni].q[1] = *(const uint4*)&Blds[row + hi * 8 + 16];   // K bkb+16..bkb+23
    }
#pragma unroll
    for (int mi = 0; mi < 2; ++mi)
#pragma unroll
      for (int ni = 0; ni < 2; ++ni)
        acc[mi][ni] = __builtin_amdgcn_wmma_f32_16x16x32_bf16(
            false, afr[mi].v, false, bfr[ni].v, (short)0, acc[mi][ni], false, false);

    __syncthreads();
  }

  // --- store D tiles: VGPR r -> M = r (+8 for upper half-wave), N = lane&15 ---
  float* outb = p.out + (long)img * p.OC * HWo;
#pragma unroll
  for (int ni = 0; ni < 2; ++ni) {
    const int col = pBlk + wn * 32 + ni * 16 + lr;
    if (col >= HWo) continue;
    float* obase = outb + col;
#pragma unroll
    for (int mi = 0; mi < 2; ++mi) {
#pragma unroll
      for (int r = 0; r < 8; ++r) {
        const int mr = mBlk + wm * 32 + mi * 16 + hi * 8 + r;
        if (mr < p.OC) obase[(long)mr * HWo] = acc[mi][ni][r];
      }
    }
  }
}

// ---------------------------------------------------------------------------
// Weight pre-conversion: f32 conv weights -> zero-padded bf16 [OCr][Kr] with
// the transposed-conv flip/transpose folded in. grid: (ceil(Kr/256), OCr).
// ---------------------------------------------------------------------------
__global__ void wcvt(const float* __restrict__ w, unsigned short* __restrict__ wb,
                     int OC, int K, int Kr, int KH, int KW, int wmode) {
  const int m = blockIdx.y;
  const int k = blockIdx.x * 256 + threadIdx.x;
  if (k >= Kr) return;
  float v = 0.f;
  if (m < OC && k < K) {
    if (wmode == 0) {
      v = w[(long)m * K + k];
    } else {
      const int KK = KH * KW;
      const int ic = k / KK;
      const int r  = k - ic * KK;
      const int kh = r / KW;
      const int kw = r - kh * KW;
      v = w[(((long)ic * OC + m) * KH + (KH - 1 - kh)) * KW + (KW - 1 - kw)];
    }
  }
  wb[(size_t)m * Kr + k] = f32_to_bf16_bits(v);
}

// ---------------------------------------------------------------------------
// Per-channel batch statistics (sum, sum of squares) for training-mode BN.
// ---------------------------------------------------------------------------
__global__ void chan_stats(const float* __restrict__ x, float* __restrict__ stats,
                           int C, int HW, int B) {
  __shared__ float rs[256], rq[256];
  const int c   = blockIdx.x;
  const int tid = threadIdx.x;
  float s = 0.f, q = 0.f;
  for (int n = 0; n < B; ++n) {
    const float* xp = x + ((long)n * C + c) * HW;
    for (int i = tid; i < HW; i += 256) {
      float v = xp[i];
      s += v; q += v * v;
    }
  }
  rs[tid] = s; rq[tid] = q;
  __syncthreads();
  for (int o = 128; o > 0; o >>= 1) {
    if (tid < o) { rs[tid] += rs[tid + o]; rq[tid] += rq[tid + o]; }
    __syncthreads();
  }
  if (tid == 0) { stats[c] = rs[0]; stats[C + c] = rq[0]; }
}

// grid: (ceil(HW/256), C, B).  mode 0: y = relu(bn(x)) ; 1: y = relu(res+bn(x))
__global__ void bn_apply(const float* __restrict__ x, const float* __restrict__ res,
                         float* __restrict__ y, const float* __restrict__ stats,
                         const float* __restrict__ g, const float* __restrict__ be,
                         int C, int HW, float invCnt, int mode) {
  const int c = blockIdx.y;
  const long base = ((long)blockIdx.z * C + c) * HW;
  const float m  = stats[c] * invCnt;
  const float v  = stats[C + c] * invCnt - m * m;
  const float sc = g[c] * rsqrtf(v + 1e-5f);
  const float sh = be[c] - m * sc;
  for (int i = blockIdx.x * 256 + threadIdx.x; i < HW; i += gridDim.x * 256) {
    float val = fmaf(x[base + i], sc, sh);
    if (mode == 1) val += res[base + i];
    y[base + i] = val > 0.f ? val : 0.f;
  }
}

// grid: (ceil(HW/256), C, B).  mode 0: y += bias ; 1: y = tanh(y + bias)
__global__ void bias_act(float* __restrict__ y, const float* __restrict__ b,
                         int C, int HW, int mode) {
  const int c = blockIdx.y;
  const long base = ((long)blockIdx.z * C + c) * HW;
  const float bc = b[c];
  for (int i = blockIdx.x * 256 + threadIdx.x; i < HW; i += gridDim.x * 256) {
    float v = y[base + i] + bc;
    if (mode == 1) v = tanhf(v);
    y[base + i] = v;
  }
}

__global__ void zero_k(float* p, long n) {
  long i = (long)blockIdx.x * blockDim.x + threadIdx.x;
  if (i < n) p[i] = 0.f;
}

// Vector quantization on the fixed 32x64x64x64 latent: nearest codebook row,
// straight-through z_q, atomic accumulation of sum((z - z_q)^2).
__global__ void vq_kernel(const float* __restrict__ ze, const float* __restrict__ cb,
                          float* __restrict__ zq, float* __restrict__ idxf,
                          float* __restrict__ lacc, int npts, int Kcodes) {
  const int pidx = blockIdx.x * blockDim.x + threadIdx.x;
  if (pidx >= npts) return;
  const int n  = pidx >> 12;          // HW = 4096
  const int hw = pidx & 4095;
  const float* zp = ze + ((long)n << 18) + hw;   // n * 64 * 4096
  float z[64];
#pragma unroll
  for (int c = 0; c < 64; ++c) z[c] = zp[c << 12];

  int best = 0;
  float bestd = 3.4e38f;
  for (int k = 0; k < Kcodes; ++k) {
    const float* e = cb + (k << 6);
    float d = 0.f;
#pragma unroll 8
    for (int c = 0; c < 64; ++c) { float t = z[c] - e[c]; d += t * t; }
    if (d < bestd) { bestd = d; best = k; }
  }
  const float* e = cb + (best << 6);
  float* o = zq + ((long)n << 18) + hw;
#pragma unroll
  for (int c = 0; c < 64; ++c) o[c << 12] = e[c];
  idxf[pidx] = (float)best;
  atomicAdd(lacc, bestd);             // bestd == sum((z - z_q)^2) for this point
}

__global__ void finalize_loss(const float* acc, float* out, float scale) {
  if (threadIdx.x == 0 && blockIdx.x == 0) out[0] = acc[0] * scale;
}

// ---------------------------------------------------------------------------
// Host orchestration
// ---------------------------------------------------------------------------
static int ilog2(int v) { int s = 0; while ((1 << s) < v) ++s; return s; }

static void launch_conv(hipStream_t s, const float* in, const float* w,
                        unsigned short* wbuf, float* out,
                        int B, int IC, int H, int W, int OC, int KH,
                        int stride, int pad, int ldil, int OH, int OW) {
  const int K     = IC * KH * KH;
  const int Kr    = (K + 31) & ~31;
  const int OCr   = (OC + 63) & ~63;
  const int wmode = (ldil == 2) ? 1 : 0;

  wcvt<<<dim3((unsigned)((Kr + 255) / 256), (unsigned)OCr), dim3(256), 0, s>>>(
      w, wbuf, OC, K, Kr, KH, KH, wmode);

  ConvP p{in, wbuf, out, IC, H, W, OC, pad, OH, OW, ilog2(OW), Kr};
  dim3 grid((unsigned)((OH * OW + 127) / 128), (unsigned)(OCr / 64), (unsigned)B);
  dim3 blk(256);
  if (KH == 4 && stride == 2) {
    conv_wmma<4, 4, 2, 1><<<grid, blk, 0, s>>>(p);          // strided 4x4 conv
  } else if (KH == 3) {
    conv_wmma<3, 3, 1, 1><<<grid, blk, 0, s>>>(p);          // 3x3 conv
  } else if (KH == 1) {
    conv_wmma<1, 1, 1, 1><<<grid, blk, 0, s>>>(p);          // 1x1 conv
  } else {
    conv_wmma<4, 4, 1, 2><<<grid, blk, 0, s>>>(p);          // transposed 4x4, s2, p1
  }
}

static void launch_bn(hipStream_t s, const float* x, const float* res, float* y,
                      float* stats, const float* g, const float* be,
                      int B, int C, int HW, int mode) {
  chan_stats<<<dim3(C), dim3(256), 0, s>>>(x, stats, C, HW, B);
  unsigned bx = (unsigned)((HW + 255) / 256);
  bn_apply<<<dim3(bx, C, B), dim3(256), 0, s>>>(x, res, y, stats, g, be, C, HW,
                                                1.0f / (float)((long)B * HW), mode);
}

static void launch_bias(hipStream_t s, float* y, const float* b, int B, int C, int HW, int mode) {
  unsigned bx = (unsigned)((HW + 255) / 256);
  bias_act<<<dim3(bx, C, B), dim3(256), 0, s>>>(y, b, C, HW, mode);
}

extern "C" void kernel_launch(void* const* d_in, const int* in_sizes, int n_in,
                              void* d_out, int out_size, void* d_ws, size_t ws_size,
                              hipStream_t stream) {
  (void)in_sizes; (void)n_in; (void)out_size; (void)ws_size;

  const float* x        = (const float*)d_in[0];
  const float* enc0_w   = (const float*)d_in[1];
  const float* enc0_g   = (const float*)d_in[3];
  const float* enc0_be  = (const float*)d_in[4];
  const float* enc1_w   = (const float*)d_in[5];
  const float* enc1_g   = (const float*)d_in[7];
  const float* enc1_be  = (const float*)d_in[8];
  const float* eres_w1  = (const float*)d_in[9];
  const float* eres_g1  = (const float*)d_in[11];
  const float* eres_be1 = (const float*)d_in[12];
  const float* eres_w2  = (const float*)d_in[13];
  const float* eres_g2  = (const float*)d_in[15];
  const float* eres_be2 = (const float*)d_in[16];
  const float* emb_w    = (const float*)d_in[17];
  const float* emb_b    = (const float*)d_in[18];
  const float* codebook = (const float*)d_in[19];
  const float* dres_w1  = (const float*)d_in[20];
  const float* dres_g1  = (const float*)d_in[22];
  const float* dres_be1 = (const float*)d_in[23];
  const float* dres_w2  = (const float*)d_in[24];
  const float* dres_g2  = (const float*)d_in[26];
  const float* dres_be2 = (const float*)d_in[27];
  const float* dec0_w   = (const float*)d_in[28];
  const float* dec0_b   = (const float*)d_in[29];
  const float* dect0_w  = (const float*)d_in[30];
  const float* dect0_g  = (const float*)d_in[32];
  const float* dect0_be = (const float*)d_in[33];
  const float* dect1_w  = (const float*)d_in[34];
  const float* dect1_b  = (const float*)d_in[35];
  // Note: biases before training-mode BN (enc0/enc1/eres/dres/dect0) cancel
  // exactly inside the BN (mean shifts by the same bias), so they are unused.

  float* out = (float*)d_out;
  float* ws  = (float*)d_ws;

  // workspace layout (floats; all offsets keep 16-B alignment)
  const long N0 = 33554432;  // 32*64*128*128
  const long N1 = 16777216;  // 32*128*64*64
  const long N2 = 16777216;
  const long N3 = 8388608;   // 32*64*64*64
  const long N4 = 8388608;
  const long N5 = 8388608;
  float* t0    = ws;
  float* t1    = t0 + N0;
  float* t2    = t1 + N1;
  float* t3    = t2 + N2;   // z_e
  float* t4    = t3 + N3;   // z_q / decoder ping
  float* t5    = t4 + N4;   // decoder pong
  float* stats = t5 + N5;   // 256 floats
  float* lacc  = stats + 256;
  unsigned short* wbuf = (unsigned short*)(stats + 512);   // 16-B aligned; 300 KB max

  const long RECON = 32L * 3 * 256 * 256;   // 6291456
  float* out_recon = out;
  float* out_loss  = out + RECON;
  float* out_idx   = out + RECON + 1;

  // ---- encoder ----
  launch_conv(stream, x, enc0_w, wbuf, t0, 32, 3, 256, 256, 64, 4, 2, 1, 1, 128, 128);
  launch_bn(stream, t0, nullptr, t0, stats, enc0_g, enc0_be, 32, 64, 128 * 128, 0);

  launch_conv(stream, t0, enc1_w, wbuf, t1, 32, 64, 128, 128, 128, 4, 2, 1, 1, 64, 64);
  launch_bn(stream, t1, nullptr, t1, stats, enc1_g, enc1_be, 32, 128, 64 * 64, 0);

  for (int i = 0; i < 2; ++i) {
    const float* w1 = eres_w1 + (long)i * 128 * 128 * 9;
    const float* w2 = eres_w2 + (long)i * 128 * 128 * 9;
    launch_conv(stream, t1, w1, wbuf, t2, 32, 128, 64, 64, 128, 3, 1, 1, 1, 64, 64);
    launch_bn(stream, t2, nullptr, t2, stats, eres_g1 + i * 128, eres_be1 + i * 128,
              32, 128, 64 * 64, 0);
    launch_conv(stream, t2, w2, wbuf, t0, 32, 128, 64, 64, 128, 3, 1, 1, 1, 64, 64);
    launch_bn(stream, t0, t1, t1, stats, eres_g2 + i * 128, eres_be2 + i * 128,
              32, 128, 64 * 64, 1);
  }

  // ---- z_e = 1x1 conv + bias ----
  launch_conv(stream, t1, emb_w, wbuf, t3, 32, 128, 64, 64, 64, 1, 1, 0, 1, 64, 64);
  launch_bias(stream, t3, emb_b, 32, 64, 64 * 64, 0);

  // ---- vector quantization ----
  zero_k<<<dim3(1), dim3(64), 0, stream>>>(lacc, 1);
  const int npts = 32 * 64 * 64;
  vq_kernel<<<dim3((npts + 255) / 256), dim3(256), 0, stream>>>(
      t3, codebook, t4, out_idx, lacc, npts, 512);
  finalize_loss<<<dim3(1), dim3(64), 0, stream>>>(
      lacc, out_loss, 0.25f / (float)(32L * 64 * 64 * 64));

  // ---- decoder ----
  for (int i = 0; i < 2; ++i) {
    const float* w1 = dres_w1 + (long)i * 64 * 64 * 9;
    const float* w2 = dres_w2 + (long)i * 64 * 64 * 9;
    launch_conv(stream, t4, w1, wbuf, t5, 32, 64, 64, 64, 64, 3, 1, 1, 1, 64, 64);
    launch_bn(stream, t5, nullptr, t5, stats, dres_g1 + i * 64, dres_be1 + i * 64,
              32, 64, 64 * 64, 0);
    launch_conv(stream, t5, w2, wbuf, t3, 32, 64, 64, 64, 64, 3, 1, 1, 1, 64, 64);
    launch_bn(stream, t3, t4, t4, stats, dres_g2 + i * 64, dres_be2 + i * 64,
              32, 64, 64 * 64, 1);
  }

  launch_conv(stream, t4, dec0_w, wbuf, t2, 32, 64, 64, 64, 128, 3, 1, 1, 1, 64, 64);
  launch_bias(stream, t2, dec0_b, 32, 128, 64 * 64, 0);

  // transposed conv (k4,s2,p1) == lhs-dilation-2, pad-2, flipped/transposed weights
  launch_conv(stream, t2, dect0_w, wbuf, t0, 32, 128, 64, 64, 64, 4, 1, 2, 2, 128, 128);
  launch_bn(stream, t0, nullptr, t0, stats, dect0_g, dect0_be, 32, 64, 128 * 128, 0);

  launch_conv(stream, t0, dect1_w, wbuf, out_recon, 32, 64, 128, 128, 3, 4, 1, 2, 2, 256, 256);
  launch_bias(stream, out_recon, dect1_b, 32, 3, 256 * 256, 1);  // bias + tanh
}
